// cnnblstm_with_adabn_7928509629133
// MI455X (gfx1250) — compile-verified, matrix operand order checked by reading, not run on hardware
//
#include <hip/hip_runtime.h>
#include <hip/hip_bf16.h>

typedef __attribute__((ext_vector_type(16))) _Float16 v16h;
typedef __attribute__((ext_vector_type(8)))  float    v8f;

// Problem constants
static constexpr int CB   = 512;   // batch
static constexpr int CT   = 800;   // input time
static constexpr int FIN  = 3;
static constexpr int NFC  = 128;   // conv channels
static constexpr int KW   = 15;
static constexpr int CH   = 150;   // hidden
static constexpr int CL   = 393;   // pooled length
static constexpr int CG   = 600;   // 4*H
static constexpr int GP   = 608;   // padded gate dim (38 * 16)
static constexpr int HP   = 160;   // padded hidden K (5 * 32)
static constexpr int KC   = 288;   // concat K = NFC + HP  (9 * 32)

// persistent recurrence kernel shape
static constexpr int NBLK = 76;    // co-resident blocks
static constexpr int NTHR = 256;   // 8 waves / block -> 608 waves total

__device__ __forceinline__ float hsig(float x) {
    return fminf(fmaxf(0.2f * x + 0.5f, 0.f), 1.f);
}

// ---------------- Conv1d(3->128,k15) + ReLU + MaxPool(2) ----------------
__global__ void k_conv(const float* __restrict__ x, const float* __restrict__ w,
                       const float* __restrict__ bias, float* __restrict__ y)
{
    int f = blockIdx.y;
    int i = blockIdx.x * blockDim.x + threadIdx.x;   // b*CL + l  (exact: 786*256)
    int b = i / CL, l = i % CL;
    __shared__ float wsm[FIN * KW];
    if (threadIdx.x < FIN * KW) wsm[threadIdx.x] = w[f * FIN * KW + threadIdx.x];
    __syncthreads();
    float a0 = bias[f], a1 = bias[f];
    const float* xb = x + (size_t)b * FIN * CT;
    #pragma unroll
    for (int c = 0; c < FIN; ++c) {
        const float* xc = xb + c * CT + 2 * l;
        #pragma unroll
        for (int k = 0; k < KW; ++k) {
            float wv = wsm[c * KW + k];
            a0 = fmaf(xc[k],     wv, a0);
            a1 = fmaf(xc[k + 1], wv, a1);
        }
    }
    y[((size_t)b * NFC + f) * CL + l] = fmaxf(fmaxf(a0, 0.f), fmaxf(a1, 0.f));
}

// ---------------- AdaBN1 stats per channel over (B, L) ----------------
__global__ void k_bn1_stats(const float* __restrict__ y, const float* __restrict__ g,
                            const float* __restrict__ be, float* scale, float* shift)
{
    int f = blockIdx.x;
    __shared__ float s1[256], s2[256];
    float a = 0.f, b = 0.f;
    for (int j = threadIdx.x; j < CB * CL; j += 256) {
        int bi = j / CL, l = j % CL;
        float v = y[((size_t)bi * NFC + f) * CL + l];
        a += v; b += v * v;
    }
    s1[threadIdx.x] = a; s2[threadIdx.x] = b; __syncthreads();
    for (int o = 128; o > 0; o >>= 1) {
        if (threadIdx.x < o) { s1[threadIdx.x] += s1[threadIdx.x + o]; s2[threadIdx.x] += s2[threadIdx.x + o]; }
        __syncthreads();
    }
    if (threadIdx.x == 0) {
        float n = (float)(CB * CL);
        float mean = s1[0] / n;
        float var  = s2[0] / n - mean * mean;
        float sc = g[f] * rsqrtf(var + 1e-5f);
        scale[f] = sc; shift[f] = be[f] - mean * sc;
    }
}

// normalize + transpose (b,f,l) -> seq (l, b, f) in fp16
__global__ void k_bn1_apply(const float* __restrict__ y, const float* __restrict__ scale,
                            const float* __restrict__ shift, _Float16* __restrict__ seq)
{
    size_t i = (size_t)blockIdx.x * blockDim.x + threadIdx.x;
    if (i >= (size_t)CL * CB * NFC) return;
    int f = (int)(i % NFC);
    int b = (int)((i / NFC) % CB);
    int l = (int)(i / ((size_t)NFC * CB));
    float v = y[((size_t)b * NFC + f) * CL + l];
    seq[i] = (_Float16)(v * scale[f] + shift[f]);
}

// ---------------- cast + concat LSTM weights: Wcat (2, GP, KC) fp16, bias (2, GP) ----------------
__global__ void k_cast_weights(const float* wihf, const float* whhf,
                               const float* wihb, const float* whhb,
                               const float* bihf, const float* bhhf,
                               const float* bihb, const float* bhhb,
                               _Float16* __restrict__ wcat, float* __restrict__ biascat)
{
    int i = blockIdx.x * blockDim.x + threadIdx.x;
    if (i >= 2 * GP * KC) return;
    int dir = i / (GP * KC);
    int r   = i % (GP * KC);
    int gI  = r / KC;
    int k   = r % KC;
    float v = 0.f;
    if (gI < CG) {
        const float* wih = dir ? wihb : wihf;
        const float* whh = dir ? whhb : whhf;
        if (k < NFC)            v = wih[(size_t)gI * NFC + k];
        else if (k < NFC + CH)  v = whh[(size_t)gI * CH + (k - NFC)];
    }
    wcat[i] = (_Float16)v;
    if (k == 0) {
        float bv = 0.f;
        if (gI < CG) bv = dir ? (bihb[gI] + bhhb[gI]) : (bihf[gI] + bhhf[gI]);
        biascat[dir * GP + gI] = bv;
    }
}

// ---------------- init h, c, running max, grid-barrier state ----------------
__global__ void k_init(_Float16* h, float* c, float* hmax, unsigned* bar)
{
    int i = blockIdx.x * blockDim.x + threadIdx.x;
    if (i < 2 * CB * HP) h[i] = (_Float16)0.f;
    if (i < 2 * CB * CH) c[i] = 0.f;
    if (i < CB * 2 * CH) hmax[i] = -1e30f;
    if (i < 2)           bar[i] = 0u;
}

// ---------------- device-wide generation barrier ----------------
__device__ __forceinline__ void gridbar(unsigned* cnt, unsigned* gen)
{
    __threadfence();            // release stores from this block
    __syncthreads();
    if (threadIdx.x == 0) {
        unsigned g = *(volatile unsigned*)gen;
        if (atomicAdd(cnt, 1u) == (unsigned)(NBLK - 1)) {
            *(volatile unsigned*)cnt = 0u;
            __threadfence();
            atomicAdd(gen, 1u);
        } else {
            while (*(volatile unsigned*)gen == g) { __builtin_amdgcn_s_sleep(2); }
        }
    }
    __syncthreads();
    __threadfence();            // acquire side
}

// ---------------- one 16(M) x 32(N) WMMA gate tile, K = 288 (9 k-steps) ----------------
__device__ __forceinline__ void gates_tile(const _Float16* __restrict__ seq,
                                           const _Float16* __restrict__ hbuf,
                                           const _Float16* __restrict__ wcat,
                                           const float* __restrict__ biascat,
                                           float* __restrict__ gates,
                                           int dir, int m0, int n0, int t, int lane)
{
    int mr = lane & 15;     // row (A) / col (B) within tile
    int hf = lane >> 4;     // K-half selector

    const _Float16* aSeq = seq  + ((size_t)t * CB + m0 + mr) * NFC;
    const _Float16* aH   = hbuf + (size_t)dir * CB * HP + (size_t)(m0 + mr) * HP;
    const _Float16* bW0  = wcat + (size_t)dir * GP * KC + (size_t)(n0 + mr) * KC;
    const _Float16* bW1  = bW0 + 16 * KC;

    v8f acc0 = {}; v8f acc1 = {};
    union Frag { uint4 u[2]; v16h h; };

    #pragma unroll
    for (int kk = 0; kk < KC / 32; ++kk) {
        const int k0 = kk * 32;
        Frag a, b0, b1;
        // A fragment: lane holds row (m0+mr); two contiguous 8-f16 runs -> 2x b128 loads
        const _Float16* ap = (k0 < NFC) ? (aSeq + k0 + 8 * hf)
                                        : (aH + (k0 - NFC) + 8 * hf);
        a.u[0] = *(const uint4*)(ap);
        a.u[1] = *(const uint4*)(ap + 16);
        const _Float16* bp0 = bW0 + k0 + 8 * hf;
        b0.u[0] = *(const uint4*)(bp0);
        b0.u[1] = *(const uint4*)(bp0 + 16);
        const _Float16* bp1 = bW1 + k0 + 8 * hf;
        b1.u[0] = *(const uint4*)(bp1);
        b1.u[1] = *(const uint4*)(bp1 + 16);
        acc0 = __builtin_amdgcn_wmma_f32_16x16x32_f16(false, a.h, false, b0.h, (short)0, acc0, false, false);
        acc1 = __builtin_amdgcn_wmma_f32_16x16x32_f16(false, a.h, false, b1.h, (short)0, acc1, false, false);
    }

    float* gOut = gates + (size_t)dir * CB * GP;
    const float* bc = biascat + dir * GP;
    int col = n0 + (lane & 15);
    #pragma unroll
    for (int r = 0; r < 8; ++r) {
        int row = m0 + r + 8 * hf;
        gOut[(size_t)row * GP + col]      = acc0[r] + bc[col];
        gOut[(size_t)row * GP + col + 16] = acc1[r] + bc[col + 16];
    }
}

// ---------------- persistent bidirectional recurrence: all 393 steps in one launch ----------------
__global__ void k_lstm_persist(const _Float16* __restrict__ seq,   // (CL*CB, NFC)
                               _Float16* __restrict__ hbuf,        // (2, CB, HP)
                               const _Float16* __restrict__ wcat,  // (2, GP, KC)
                               const float* __restrict__ biascat,  // (2, GP)
                               float* __restrict__ gates,          // (2, CB, GP)
                               float* __restrict__ cbuf,           // (2, CB, CH)
                               float* __restrict__ hmax,           // (CB, 2*CH)
                               unsigned* __restrict__ bar)
{
    const int lane    = threadIdx.x & 31;
    const int waveId  = (blockIdx.x * NTHR + threadIdx.x) >> 5;      // 0..607
    const int nWaves  = NBLK * NTHR / 32;                            // 608
    const int tilesTotal = 2 * (CB / 16) * (GP / 32);                // 1216
    const int tid     = blockIdx.x * NTHR + threadIdx.x;
    unsigned* cnt = bar;
    unsigned* gen = bar + 1;

    for (int s = 0; s < CL; ++s) {
        // ---- phase 1: gates = [seq_t | h] @ Wcat^T + bias (WMMA) ----
        // each wave owns the SAME 2 tiles every step -> W fragments stay WGP$-resident
        for (int tile = waveId; tile < tilesTotal; tile += nWaves) {
            int dir = tile / (tilesTotal / 2);
            int rem = tile % (tilesTotal / 2);
            int m0  = (rem / (GP / 32)) * 16;
            int n0  = (rem % (GP / 32)) * 32;
            int t   = (dir == 0) ? s : (CL - 1 - s);
            gates_tile(seq, hbuf, wcat, biascat, gates, dir, m0, n0, t, lane);
        }
        gridbar(cnt, gen);

        // ---- phase 2: elementwise LSTM update + running time-max ----
        for (int i = tid; i < 2 * CB * CH; i += NBLK * NTHR) {
            int dir = i / (CB * CH);
            int r   = i % (CB * CH);
            int b   = r / CH;
            int j   = r % CH;
            const float* gt = gates + (size_t)dir * CB * GP + (size_t)b * GP;
            float ig = hsig(gt[j]);
            float fg = hsig(gt[CH + j]);
            float gg = tanhf(gt[2 * CH + j]);
            float og = hsig(gt[3 * CH + j]);
            float* cp = cbuf + (size_t)dir * CB * CH + (size_t)b * CH + j;
            float cv = fg * (*cp) + ig * gg;
            *cp = cv;
            float hv = og * tanhf(cv);
            hbuf[(size_t)dir * CB * HP + (size_t)b * HP + j] = (_Float16)hv;
            float* hm = hmax + (size_t)b * (2 * CH) + dir * CH + j;
            *hm = fmaxf(*hm, hv);
        }
        gridbar(cnt, gen);
    }
}

// ---------------- per-column AdaBN stats over batch (rows = CB) ----------------
__global__ void k_colstats(const float* __restrict__ src, int cols,
                           const float* __restrict__ g, const float* __restrict__ be,
                           float* scale, float* shift)
{
    int cI = blockIdx.x;
    __shared__ float s1[256], s2[256];
    float a = 0.f, b = 0.f;
    for (int r = threadIdx.x; r < CB; r += 256) {
        float v = src[(size_t)r * cols + cI];
        a += v; b += v * v;
    }
    s1[threadIdx.x] = a; s2[threadIdx.x] = b; __syncthreads();
    for (int o = 128; o > 0; o >>= 1) {
        if (threadIdx.x < o) { s1[threadIdx.x] += s1[threadIdx.x + o]; s2[threadIdx.x] += s2[threadIdx.x + o]; }
        __syncthreads();
    }
    if (threadIdx.x == 0) {
        float mean = s1[0] / (float)CB;
        float var  = s2[0] / (float)CB - mean * mean;
        float sc = g[cI] * rsqrtf(var + 1e-5f);
        scale[cI] = sc; shift[cI] = be[cI] - mean * sc;
    }
}

// ---------------- FC1 (300 -> 50) with on-the-fly AdaBN, ReLU ----------------
__global__ void k_fc1(const float* __restrict__ pooled, const float* __restrict__ scale2,
                      const float* __restrict__ shift2, const float* __restrict__ w1,
                      const float* __restrict__ b1, float* __restrict__ z1)
{
    int i = blockIdx.x * blockDim.x + threadIdx.x;
    if (i >= CB * 50) return;
    int b = i / 50, o = i % 50;
    float a = b1[o];
    for (int cI = 0; cI < 2 * CH; ++cI)
        a = fmaf(pooled[(size_t)b * (2 * CH) + cI] * scale2[cI] + shift2[cI], w1[o * (2 * CH) + cI], a);
    z1[i] = fmaxf(a, 0.f);
}

// ---------------- FC2 (50 -> 10) with on-the-fly AdaBN + softmax ----------------
__global__ void k_head2(const float* __restrict__ z1, const float* __restrict__ scale3,
                        const float* __restrict__ shift3, const float* __restrict__ w2,
                        const float* __restrict__ b2, float* __restrict__ out)
{
    int b = blockIdx.x * blockDim.x + threadIdx.x;
    if (b >= CB) return;
    float zn[50];
    #pragma unroll
    for (int cI = 0; cI < 50; ++cI) zn[cI] = z1[(size_t)b * 50 + cI] * scale3[cI] + shift3[cI];
    float lg[10]; float mx = -1e30f;
    #pragma unroll
    for (int o = 0; o < 10; ++o) {
        float a = b2[o];
        for (int cI = 0; cI < 50; ++cI) a = fmaf(zn[cI], w2[o * 50 + cI], a);
        lg[o] = a; mx = fmaxf(mx, a);
    }
    float sum = 0.f;
    #pragma unroll
    for (int o = 0; o < 10; ++o) { lg[o] = __expf(lg[o] - mx); sum += lg[o]; }
    float inv = 1.f / sum;
    #pragma unroll
    for (int o = 0; o < 10; ++o) out[(size_t)b * 10 + o] = lg[o] * inv;
}

extern "C" void kernel_launch(void* const* d_in, const int* in_sizes, int n_in,
                              void* d_out, int out_size, void* d_ws, size_t ws_size,
                              hipStream_t stream)
{
    const float* x      = (const float*)d_in[0];
    const float* conv_w = (const float*)d_in[1];
    const float* conv_b = (const float*)d_in[2];
    const float* bn1_g  = (const float*)d_in[3];
    const float* bn1_b  = (const float*)d_in[4];
    const float* w_ih_f = (const float*)d_in[5];
    const float* w_hh_f = (const float*)d_in[6];
    const float* b_ih_f = (const float*)d_in[7];
    const float* b_hh_f = (const float*)d_in[8];
    const float* w_ih_b = (const float*)d_in[9];
    const float* w_hh_b = (const float*)d_in[10];
    const float* b_ih_b = (const float*)d_in[11];
    const float* b_hh_b = (const float*)d_in[12];
    const float* bn2_g  = (const float*)d_in[13];
    const float* bn2_b  = (const float*)d_in[14];
    const float* fc1_w  = (const float*)d_in[15];
    const float* fc1_b  = (const float*)d_in[16];
    const float* bn3_g  = (const float*)d_in[17];
    const float* bn3_b  = (const float*)d_in[18];
    const float* fc2_w  = (const float*)d_in[19];
    const float* fc2_b  = (const float*)d_in[20];
    float* out = (float*)d_out;

    // bump allocator over workspace
    char* p = (char*)d_ws;
    auto alloc = [&](size_t bytes) -> void* {
        void* r = (void*)p;
        p += (bytes + 255) & ~(size_t)255;
        return r;
    };
    float*    y       = (float*)   alloc((size_t)CB * NFC * CL * 4);       // conv output
    float*    scale1  = (float*)   alloc(NFC * 4);
    float*    shift1  = (float*)   alloc(NFC * 4);
    _Float16* seq     = (_Float16*)alloc((size_t)CL * CB * NFC * 2);       // (t,b,f) fp16
    _Float16* wcat    = (_Float16*)alloc((size_t)2 * GP * KC * 2);
    float*    biascat = (float*)   alloc(2 * GP * 4);
    _Float16* hbuf    = (_Float16*)alloc((size_t)2 * CB * HP * 2);
    float*    cbuf    = (float*)   alloc((size_t)2 * CB * CH * 4);
    float*    hmax    = (float*)   alloc((size_t)CB * 2 * CH * 4);
    float*    gates   = (float*)   alloc((size_t)2 * CB * GP * 4);
    unsigned* bar     = (unsigned*)alloc(2 * 4);
    float*    scale2  = (float*)   alloc(2 * CH * 4);
    float*    shift2  = (float*)   alloc(2 * CH * 4);
    float*    z1      = (float*)   alloc((size_t)CB * 50 * 4);
    float*    scale3  = (float*)   alloc(50 * 4);
    float*    shift3  = (float*)   alloc(50 * 4);

    // Conv + ReLU + MaxPool
    k_conv<<<dim3((CB * CL) / 256, NFC), 256, 0, stream>>>(x, conv_w, conv_b, y);
    // AdaBN1
    k_bn1_stats<<<NFC, 256, 0, stream>>>(y, bn1_g, bn1_b, scale1, shift1);
    {
        size_t n = (size_t)CL * CB * NFC;
        k_bn1_apply<<<(unsigned)((n + 255) / 256), 256, 0, stream>>>(y, scale1, shift1, seq);
    }
    // LSTM weight prep + state init (also resets barrier counters every call)
    k_cast_weights<<<(2 * GP * KC + 255) / 256, 256, 0, stream>>>(
        w_ih_f, w_hh_f, w_ih_b, w_hh_b, b_ih_f, b_hh_f, b_ih_b, b_hh_b, wcat, biascat);
    k_init<<<(2 * CB * HP + 255) / 256, 256, 0, stream>>>(hbuf, cbuf, hmax, bar);

    // Bidirectional recurrence: single persistent launch, 393 steps internally
    k_lstm_persist<<<NBLK, NTHR, 0, stream>>>(seq, hbuf, wcat, biascat, gates, cbuf, hmax, bar);

    // Head: AdaBN2 -> FC1+ReLU -> AdaBN3 -> FC2 + softmax
    k_colstats<<<2 * CH, 256, 0, stream>>>(hmax, 2 * CH, bn2_g, bn2_b, scale2, shift2);
    k_fc1<<<(CB * 50 + 255) / 256, 256, 0, stream>>>(hmax, scale2, shift2, fc1_w, fc1_b, z1);
    k_colstats<<<50, 256, 0, stream>>>(z1, 50, bn3_g, bn3_b, scale3, shift3);
    k_head2<<<(CB + 255) / 256, 256, 0, stream>>>(z1, scale3, shift3, fc2_w, fc2_b, out);
}
